// Attention_42657615184392
// MI455X (gfx1250) — matrix operand order repaired and checked by reading, not compiled, not run on hardware
//
#include <hip/hip_runtime.h>
#include <hip/hip_bf16.h>

// Flash attention for [4,16,2048,128] fp32 on gfx1250 (MI455X).
// Both GEMMs via v_wmma_f32_16x16x32_f16 (f16 in, f32 accumulate).
// One wave = one 16-row query tile; block = 8 waves = 128 query rows.
// Softmax in log2 domain (native v_exp_f32 = 2^x), fused DPP-max reductions,
// deferred row-sum reduction, vote-guarded rescale, and double-buffered
// LDS K/V tiles (one barrier per iteration, global loads hidden by compute).

typedef __attribute__((ext_vector_type(16))) _Float16 v16h;
typedef __attribute__((ext_vector_type(8)))  _Float16 v8h;
typedef __attribute__((ext_vector_type(2)))  _Float16 v2h;
typedef __attribute__((ext_vector_type(8)))  float    v8f;

#define B_   4
#define H_   16
#define S_   2048
#define D_   128
#define KT   32          // keys per iteration
#define NIT  (S_ / KT)   // 64 iterations
#define KST  136         // Ks row stride (halves), 272B = 17*16B
#define VST  40          // Vt row stride (halves), 80B  = 5*16B
#define PST  40          // P  row stride (halves), 80B  = 5*16B

// Raw max: single v_max_num_f32, no canonicalize ops (inputs are finite here).
__device__ __forceinline__ float fmaxr(float a, float b) {
  float r;
  asm("v_max_num_f32_e32 %0, %1, %2" : "=v"(r) : "v"(a), "v"(b));
  return r;
}
// Fused shuffle+max: one v_max_num_f32_dpp per butterfly step (row_xmask within
// each 16-lane row); avoids the mov_dpp + canonicalize pair fmaxf generates.
__device__ __forceinline__ float rmax16(float x) {
  float r;
  asm("v_max_num_f32_dpp %0, %1, %1 row_xmask:1 row_mask:0xf bank_mask:0xf bound_ctrl:1"
      : "=v"(r) : "v"(x));
  asm("v_max_num_f32_dpp %0, %1, %1 row_xmask:2 row_mask:0xf bank_mask:0xf bound_ctrl:1"
      : "=v"(x) : "v"(r));
  asm("v_max_num_f32_dpp %0, %1, %1 row_xmask:4 row_mask:0xf bank_mask:0xf bound_ctrl:1"
      : "=v"(r) : "v"(x));
  asm("v_max_num_f32_dpp %0, %1, %1 row_xmask:8 row_mask:0xf bank_mask:0xf bound_ctrl:1"
      : "=v"(x) : "v"(r));
  return x;
}
// Sum reduction (epilogue only): DPP mov + add is fine off the hot path.
template <int CTRL>
__device__ __forceinline__ float dpp_movf(float x) {
  int i = __builtin_bit_cast(int, x);
  i = __builtin_amdgcn_update_dpp(i, i, CTRL, 0xf, 0xf, true);
  return __builtin_bit_cast(float, i);
}
__device__ __forceinline__ float rsum16(float x) {
  x += dpp_movf<0x161>(x);
  x += dpp_movf<0x162>(x);
  x += dpp_movf<0x164>(x);
  x += dpp_movf<0x168>(x);
  return x;
}

__global__ __launch_bounds__(256) void Attention_42657615184392_kernel(
    const float* __restrict__ Q, const float* __restrict__ K,
    const float* __restrict__ V, float* __restrict__ O)
{
  __shared__ __align__(16) _Float16 Ks[2][KT * KST];     // K tiles  [32 keys][128 d]
  __shared__ __align__(16) _Float16 Vt[2][D_ * VST];     // V^T tiles [128 d][32 keys]
  __shared__ __align__(16) _Float16 Pb[8 * 16 * PST];    // per-wave P staging [16][32]

  const int tid   = threadIdx.x;
  const int wave  = tid >> 5;
  const int lane  = tid & 31;
  const int n     = lane & 15;     // column / row-in-tile index
  const int hi    = lane >> 4;     // lane half
  const int koffA = hi * 16;       // A-fragment K split: 16 contiguous
  const int koffB = hi * 16;       // B-fragment K split: 16 contiguous

  const int bh    = blockIdx.x >> 4;        // 0..63  (b*16+h)
  const int qblk  = blockIdx.x & 15;        // 0..15
  const size_t base = (size_t)bh * (S_ * D_);
  const int qrow0 = qblk * 128 + wave * 16; // this wave's query tile

  // staging assignment: each thread owns 2 adjacent keys x 8 consecutive d,
  // so transposed-V element pairs are contiguous (packed b32 stores)
  const int skey2 = (tid >> 4) * 2;   // 0,2,...,30
  const int sd8   = (tid & 15) * 8;   // 0..120

  // 1/sqrt(128) * log2(e): scores in log2 domain -> bare v_exp_f32 everywhere
  const float scale = 0.08838834764831845f * 1.4426950408889634f;

  // ---- Q A-fragments: 4 fragments of 16x32 covering d = 0..127 (scale folded)
  v16h qf[4];
  {
    const float* qp = Q + base + (size_t)(qrow0 + n) * D_;
#pragma unroll
    for (int s = 0; s < 4; ++s) {
      const float* p = qp + s * 32 + koffA;
#pragma unroll
      for (int j = 0; j < 8; ++j) qf[s][j]     = (_Float16)(p[j]      * scale);
#pragma unroll
      for (int j = 0; j < 8; ++j) qf[s][8 + j] = (_Float16)(p[8 + j] * scale);
    }
  }

  float4 kr[4], vr[4];
  auto loadRegs = [&](int tile) {
    const float* kp = K + base + (size_t)(tile * KT + skey2) * D_ + sd8;
    const float* vp = V + base + (size_t)(tile * KT + skey2) * D_ + sd8;
    kr[0] = *(const float4*)(kp);       kr[1] = *(const float4*)(kp + 4);
    kr[2] = *(const float4*)(kp + D_);  kr[3] = *(const float4*)(kp + D_ + 4);
    vr[0] = *(const float4*)(vp);       vr[1] = *(const float4*)(vp + 4);
    vr[2] = *(const float4*)(vp + D_);  vr[3] = *(const float4*)(vp + D_ + 4);
  };
  auto storeBuf = [&](int b) {
    const float k0[8] = {kr[0].x,kr[0].y,kr[0].z,kr[0].w, kr[1].x,kr[1].y,kr[1].z,kr[1].w};
    const float k1[8] = {kr[2].x,kr[2].y,kr[2].z,kr[2].w, kr[3].x,kr[3].y,kr[3].z,kr[3].w};
    const float v0[8] = {vr[0].x,vr[0].y,vr[0].z,vr[0].w, vr[1].x,vr[1].y,vr[1].z,vr[1].w};
    const float v1[8] = {vr[2].x,vr[2].y,vr[2].z,vr[2].w, vr[3].x,vr[3].y,vr[3].z,vr[3].w};
    v8h a0, a1;
#pragma unroll
    for (int j = 0; j < 4; ++j) {
      const v2h p0 = __builtin_bit_cast(v2h, __builtin_amdgcn_cvt_pkrtz(k0[2*j], k0[2*j+1]));
      const v2h p1 = __builtin_bit_cast(v2h, __builtin_amdgcn_cvt_pkrtz(k1[2*j], k1[2*j+1]));
      a0[2*j] = p0[0]; a0[2*j+1] = p0[1];
      a1[2*j] = p1[0]; a1[2*j+1] = p1[1];
    }
    *(v8h*)&Ks[b][ skey2      * KST + sd8] = a0;
    *(v8h*)&Ks[b][(skey2 + 1) * KST + sd8] = a1;
#pragma unroll
    for (int j = 0; j < 8; ++j) {
      const v2h pv = __builtin_bit_cast(v2h, __builtin_amdgcn_cvt_pkrtz(v0[j], v1[j]));
      *(v2h*)&Vt[b][(sd8 + j) * VST + skey2] = pv;   // Vt[d][key0], Vt[d][key1]
    }
  };

  // ---- software pipeline prologue: tile 0 -> LDS buf0, tile 1 -> registers
  loadRegs(0);
  storeBuf(0);
  loadRegs(1);

  // ---- accumulators: O = 16x128 -> 8 C-tiles; running row max m, PER-LANE
  //      partial row sum l (reduced across lanes only once at the end)
  v8f o[8];
  float m[8], l[8];
#pragma unroll
  for (int dt = 0; dt < 8; ++dt) o[dt] = (v8f){0.f,0.f,0.f,0.f,0.f,0.f,0.f,0.f};
#pragma unroll
  for (int g = 0; g < 8; ++g) { m[g] = -1e30f; l[g] = 0.f; }

  for (int it = 0; it < NIT; ++it) {
    __syncthreads();  // buf[it&1] staged by all waves; previous readers done
    const _Float16* ksb = &Ks[it & 1][0];
    const _Float16* vtb = &Vt[it & 1][0];

    // ---- S = Q K^T : two 16x16 tiles (keys 0..15 and 16..31), K-dim = 128
    v8f s0 = (v8f){0.f,0.f,0.f,0.f,0.f,0.f,0.f,0.f};
    v8f s1 = s0;
#pragma unroll
    for (int ds = 0; ds < 4; ++ds) {
      v16h b0, b1;
      const _Float16* p0 = &ksb[n * KST        + ds * 32 + koffB]; // col = key n
      const _Float16* p1 = &ksb[(16 + n) * KST + ds * 32 + koffB]; // col = key 16+n
#pragma unroll
      for (int j = 0; j < 16; ++j) { b0[j] = p0[j]; b1[j] = p1[j]; }
      s0 = __builtin_amdgcn_wmma_f32_16x16x32_f16(false, qf[ds], false, b0,
                                                  (short)0, s0, false, false);
      s1 = __builtin_amdgcn_wmma_f32_16x16x32_f16(false, qf[ds], false, b1,
                                                  (short)0, s1, false, false);
    }

    // ---- online softmax, log2 domain (C layout: row = g + 8*hi)
    float mn[8];
    bool upd = false;
#pragma unroll
    for (int g = 0; g < 8; ++g) {
      const float mx = rmax16(fmaxr(s0[g], s1[g]));
      mn[g] = fmaxr(m[g], mx);
      upd = upd || (mn[g] > m[g]);
    }
    if (__any(upd)) {  // max moved for some row: rescale O and l (alpha = 2^(m-mn))
#pragma unroll
      for (int g = 0; g < 8; ++g) {
        const float a = __builtin_amdgcn_exp2f(m[g] - mn[g]);
        m[g] = mn[g];
        l[g] *= a;
#pragma unroll
        for (int dt = 0; dt < 8; ++dt) o[dt][g] *= a;
      }
    }
#pragma unroll
    for (int g = 0; g < 8; ++g) {
      const float x0 = __builtin_amdgcn_exp2f(s0[g] - m[g]);
      const float x1 = __builtin_amdgcn_exp2f(s1[g] - m[g]);
      s0[g] = x0; s1[g] = x1;
      l[g] += x0 + x1;          // per-lane partial; reduced after the key loop
    }

    // ---- P: C layout -> LDS (f16, row major) -> A-fragment reload
    _Float16* pw = &Pb[wave * (16 * PST)];
#pragma unroll
    for (int g = 0; g < 8; ++g) {
      pw[(g + 8 * hi) * PST + n]      = (_Float16)s0[g];
      pw[(g + 8 * hi) * PST + 16 + n] = (_Float16)s1[g];
    }
    // same-wave DS ordering; drain DScnt before cross-lane reload
    asm volatile("s_wait_dscnt 0" ::: "memory");
    v16h pa;
    {
      const _Float16* pr = &pw[n * PST + koffA];
#pragma unroll
      for (int j = 0; j < 8; ++j) pa[j]     = pr[j];
#pragma unroll
      for (int j = 0; j < 8; ++j) pa[8 + j] = pr[8 + j];
    }

    // ---- O += P V : 8 output tiles across d, single K=32 WMMA each
#pragma unroll
    for (int dt = 0; dt < 8; ++dt) {
      v16h vb;
      const _Float16* vp = &vtb[(dt * 16 + n) * VST + koffB]; // col = d-col dt*16+n
#pragma unroll
      for (int j = 0; j < 16; ++j) vb[j] = vp[j];
      o[dt] = __builtin_amdgcn_wmma_f32_16x16x32_f16(false, pa, false, vb,
                                                     (short)0, o[dt], false, false);
    }

    // ---- stage next tile into the other buffer, start tile it+2 loads
    if (it + 1 < NIT) {
      storeBuf((it + 1) & 1);
      loadRegs(it + 2 < NIT ? it + 2 : it + 1);
    }
  }

  // ---- epilogue: reduce per-lane l partials across the row, normalize, store
#pragma unroll
  for (int g = 0; g < 8; ++g) {
    const float inv = 1.0f / rsum16(l[g]);
    float* op = O + base + (size_t)(qrow0 + g + 8 * hi) * D_ + n;
#pragma unroll
    for (int dt = 0; dt < 8; ++dt) op[dt * 16] = o[dt][g] * inv;
  }
}

extern "C" void kernel_launch(void* const* d_in, const int* in_sizes, int n_in,
                              void* d_out, int out_size, void* d_ws, size_t ws_size,
                              hipStream_t stream) {
  (void)in_sizes; (void)n_in; (void)out_size; (void)d_ws; (void)ws_size;
  const float* Q = (const float*)d_in[0];
  const float* K = (const float*)d_in[1];
  const float* V = (const float*)d_in[2];
  float* O = (float*)d_out;
  // 64 (b,h) pairs x 16 query blocks of 128 rows = 1024 blocks, 256 threads (8 waves)
  dim3 grid(B_ * H_ * (S_ / 128));
  Attention_42657615184392_kernel<<<grid, dim3(256), 0, stream>>>(Q, K, V, O);
}
